// MultiMamba_66503273611840
// MI455X (gfx1250) — compile-verified, matrix operand order checked by reading, not run on hardware
//
#include <hip/hip_runtime.h>
#include <hip/hip_bf16.h>
#include <cmath>

// ---------------- problem constants ----------------
#define BATCH   32
#define HH      28
#define WW      28
#define LL      (HH*WW)          // 784
#define DMODEL  384
#define DIN     768              // D_MODEL * EXPAND
#define NST     16
#define RRANK   24               // ceil(384/16)
#define KCONV   4
#define NDIR    4
#define RC      96               // int(DIN*0.125)
#define XDBL_W  (RRANK + 2*NST)  // 56
#define XDBL_P  64               // padded N for x_proj GEMM
#define RRANK_P 32               // padded K for dt GEMM
#define MROWS   (BATCH*LL)       // 25088

typedef __attribute__((ext_vector_type(16))) __bf16 v16bf;
typedef __attribute__((ext_vector_type(8)))  __bf16 bf16x8;
typedef __attribute__((ext_vector_type(8)))  float  v8f;

// map sequence index t -> spatial index l for each scan direction
__device__ __forceinline__ int map_pos(int dir, int t) {
    switch (dir) {
        case 0:  return t;                                   // h
        case 1:  return LL - 1 - t;                          // h_flip
        case 2:  { int w = t / HH, h = t % HH; return h*WW + w; }        // v (col-major)
        default: { int tt = LL-1-t; int w = tt / HH, h = tt % HH; return h*WW + w; } // v_flip
    }
}

__device__ __forceinline__ float sigmoidf_(float x) { return 1.0f / (1.0f + __expf(-x)); }
__device__ __forceinline__ float siluf_(float x)    { return x * sigmoidf_(x); }
__device__ __forceinline__ float softplusf_(float x){ return (x > 20.0f) ? x : log1pf(__expf(x)); }

// One lane's 16x32 bf16 fragment row.  CDNA5 16-bit operand layout:
// lanes 0-15 hold K {kk..kk+7, kk+16..kk+23}, lanes 16-31 hold
// K {kk+8..kk+15, kk+24..kk+31}; VGPR p holds K pair (2p,2p+1).
// With bf16-resident operands each run of 8 is one global_load_b128.
__device__ __forceinline__ v16bf load_frag(const __bf16* __restrict__ p /* row + kk + half*8 */)
{
    bf16x8 lo = *(const bf16x8*)(p);
    bf16x8 hi = *(const bf16x8*)(p + 16);
    v16bf f;
    #pragma unroll
    for (int j = 0; j < 8; ++j) { f[j] = lo[j]; f[8 + j] = hi[j]; }
    return f;
}

// =====================================================================
// GEMM: C[M][Nreal] = A[M][K] * W[Npad][K]^T.  bf16 in, fp32 WMMA acc.
// Host guarantees: M%16==0, K%32==0, W has Npad = gridDim.x*64 rows
// (zero padded).  One wave32 computes a 16x64 tile: the A fragment is
// reused across 4 v_wmma accumulators.  EXEC all-ones throughout; only
// 10x b128 loads per 4 WMMAs.
// =====================================================================
__global__ __launch_bounds__(32)
void gemm_wmma(const __bf16* __restrict__ A, int lda,
               const __bf16* __restrict__ W, int ldw,
               float* __restrict__ C, int ldc,
               int Nreal, int K)
{
    const int lane = threadIdx.x;          // 0..31
    const int half = lane >> 4;            // 0|1
    const int lr   = lane & 15;            // 0..15
    const int m0   = blockIdx.y * 16;
    const int n0   = blockIdx.x * 64;

    const __bf16* arow = A + (size_t)(m0 + lr) * lda + half * 8;
    const __bf16* brow = W + (size_t)(n0 + lr) * ldw + half * 8;

    v8f acc0 = {}, acc1 = {}, acc2 = {}, acc3 = {};

    for (int kk = 0; kk < K; kk += 32) {
        if (kk + 32 < K) {
            __builtin_prefetch(arow + kk + 32, 0, 0);
            __builtin_prefetch(brow + kk + 32, 0, 0);
        }
        const v16bf af = load_frag(arow + kk);
        const v16bf b0 = load_frag(brow + kk);
        const v16bf b1 = load_frag(brow + (size_t)16 * ldw + kk);
        const v16bf b2 = load_frag(brow + (size_t)32 * ldw + kk);
        const v16bf b3 = load_frag(brow + (size_t)48 * ldw + kk);
        acc0 = __builtin_amdgcn_wmma_f32_16x16x32_bf16(false, af, false, b0, (short)0, acc0, false, false);
        acc1 = __builtin_amdgcn_wmma_f32_16x16x32_bf16(false, af, false, b1, (short)0, acc1, false, false);
        acc2 = __builtin_amdgcn_wmma_f32_16x16x32_bf16(false, af, false, b2, (short)0, acc2, false, false);
        acc3 = __builtin_amdgcn_wmma_f32_16x16x32_bf16(false, af, false, b3, (short)0, acc3, false, false);
    }

    // C/D layout: VGPR r -> M = m0 + r + half*8 ; N = lr (+16 per sub-tile)
    #pragma unroll
    for (int s = 0; s < 4; ++s) {
        const v8f acc = (s == 0) ? acc0 : (s == 1) ? acc1 : (s == 2) ? acc2 : acc3;
        const int n = n0 + s * 16 + lr;
        if (n < Nreal) {
            #pragma unroll
            for (int r = 0; r < 8; ++r) {
                const int m = m0 + r + half * 8;
                C[(size_t)m * ldc + n] = acc[r];
            }
        }
    }
}

// cast fp32 -> bf16 with optional row/col zero padding
// dst[r*cols_dst + c] = (r<rows_src && c<cols_src) ? bf16(src[r*src_stride + c]) : 0
__global__ void cast_pad_bf16(const float* __restrict__ src, __bf16* __restrict__ dst,
                              int rows_src, int cols_src, int src_stride,
                              int rows_dst, int cols_dst)
{
    const size_t i = (size_t)blockIdx.x * blockDim.x + threadIdx.x;
    if (i >= (size_t)rows_dst * cols_dst) return;
    const int r = (int)(i / cols_dst), c = (int)(i - (size_t)r * cols_dst);
    dst[i] = (r < rows_src && c < cols_src) ? (__bf16)src[(size_t)r * src_stride + c]
                                            : (__bf16)0.0f;
}

// =====================================================================
// Depthwise causal conv (K=4) + SiLU along the direction-mapped sequence.
// Emits both the fp32 result (for the scan) and its bf16 copy (A operand
// of the x_proj GEMM).
// =====================================================================
__global__ void conv_silu_kernel(const float* __restrict__ xz,
                                 const float* __restrict__ conv_w,  // (DIN,4)
                                 const float* __restrict__ conv_b,  // (DIN)
                                 float* __restrict__ xconv,
                                 __bf16* __restrict__ xconv_bf,
                                 int dir)
{
    const int idx = blockIdx.x * blockDim.x + threadIdx.x;   // t*DIN + d
    if (idx >= LL * DIN) return;
    const int b = blockIdx.y;
    const int t = idx / DIN;
    const int d = idx - t * DIN;

    float acc = conv_b[d];
    #pragma unroll
    for (int j = 0; j < KCONV; ++j) {
        const int ts = t - (KCONV - 1) + j;
        if (ts >= 0) {
            const int l = map_pos(dir, ts);
            acc += conv_w[d * KCONV + j] * xz[((size_t)b * LL + l) * (2*DIN) + d];
        }
    }
    const float v = siluf_(acc);
    const size_t o = ((size_t)b * LL + t) * DIN + d;
    xconv[o]    = v;
    xconv_bf[o] = (__bf16)v;
}

// =====================================================================
// Selective scan: one lane owns one (b,d) chain; 16-state recurrence in
// registers; fuses softplus(dt)+dt_b and the SiLU(z) output gate; writes
// y back at the spatial position (implements _rev_dir implicitly).
// =====================================================================
__global__ __launch_bounds__(256)
void scan_kernel(const float* __restrict__ xconv,   // (B*L, DIN) seq order
                 const float* __restrict__ delta,   // (B*L, DIN) seq order (pre dt_b/softplus)
                 const float* __restrict__ xdbl,    // (B*L, 56)  seq order
                 const float* __restrict__ xz,      // (B*L, 1536) spatial order (for z)
                 const float* __restrict__ A_log,   // (DIN,16)
                 const float* __restrict__ dt_b,    // (DIN)
                 const float* __restrict__ Dp,      // (DIN)
                 float* __restrict__ ydir,          // (B*L, DIN) spatial order
                 int dir)
{
    const int d = blockIdx.x * blockDim.x + threadIdx.x;
    if (d >= DIN) return;
    const int b = blockIdx.y;

    float Aneg[NST];
    #pragma unroll
    for (int n = 0; n < NST; ++n) Aneg[n] = -__expf(A_log[d * NST + n]);
    const float dtb = dt_b[d];
    const float dpp = Dp[d];

    float h[NST];
    #pragma unroll
    for (int n = 0; n < NST; ++n) h[n] = 0.0f;

    const float* xc_b = xconv + (size_t)b * LL * DIN;
    const float* de_b = delta + (size_t)b * LL * DIN;
    const float* xd_b = xdbl  + (size_t)b * LL * XDBL_W;

    for (int t = 0; t < LL; ++t) {
        const float u  = xc_b[(size_t)t * DIN + d];
        const float dt = softplusf_(de_b[(size_t)t * DIN + d] + dtb);
        const float du = dt * u;
        const float* bc = xd_b + (size_t)t * XDBL_W + RRANK;
        float y = 0.0f;
        #pragma unroll
        for (int n = 0; n < NST; ++n) {
            const float Bn = bc[n];
            const float Cn = bc[NST + n];
            h[n] = __expf(dt * Aneg[n]) * h[n] + du * Bn;
            y += h[n] * Cn;
        }
        y += dpp * u;
        const int l = map_pos(dir, t);
        const float z = xz[((size_t)b * LL + l) * (2*DIN) + DIN + d];
        ydir[((size_t)b * LL + l) * DIN + d] = y * siluf_(z);
    }
}

// =====================================================================
// Bi-attention stage 1: per (b,l) layernorm stats over DIN channels,
// LDS tree reduction.
// =====================================================================
__global__ __launch_bounds__(256)
void ln_stats_kernel(const float* __restrict__ y,   // (B*L, DIN)
                     float* __restrict__ mu,        // (B*L)
                     float* __restrict__ rstd)      // (B*L)
{
    __shared__ float ssum[256];
    __shared__ float ssq[256];
    const int l = blockIdx.x, b = blockIdx.y, tid = threadIdx.x;
    const float* row = y + ((size_t)b * LL + l) * DIN;
    float s = 0.0f, q = 0.0f;
    for (int d = tid; d < DIN; d += 256) { float v = row[d]; s += v; q += v * v; }
    ssum[tid] = s; ssq[tid] = q;
    __syncthreads();
    for (int off = 128; off > 0; off >>= 1) {
        if (tid < off) { ssum[tid] += ssum[tid + off]; ssq[tid] += ssq[tid + off]; }
        __syncthreads();
    }
    if (tid == 0) {
        const float m = ssum[0] * (1.0f / DIN);
        const float v = ssq[0] * (1.0f / DIN) - m * m;
        mu  [(size_t)b * LL + l] = m;
        rstd[(size_t)b * LL + l] = rsqrtf(v + 1e-5f);
    }
}

// Stage 2: g[b][d] = ln_g[d] * mean_l((y-mu)*rstd) + ln_b[d]
__global__ void ln_mean_kernel(const float* __restrict__ y,
                               const float* __restrict__ mu,
                               const float* __restrict__ rstd,
                               const float* __restrict__ ln_g,
                               const float* __restrict__ ln_b,
                               float* __restrict__ g)   // (B, DIN)
{
    const int d = blockIdx.x * blockDim.x + threadIdx.x;
    if (d >= DIN) return;
    const int b = blockIdx.y;
    float s = 0.0f;
    for (int l = 0; l < LL; ++l)
        s += (y[((size_t)b * LL + l) * DIN + d] - mu[(size_t)b * LL + l]) * rstd[(size_t)b * LL + l];
    g[(size_t)b * DIN + d] = ln_g[d] * (s * (1.0f / LL)) + ln_b[d];
}

// Stage 3: gr = gelu(g @ gr_w^T + gr_b) (RC=96) ; attn = sigmoid(gr @ cs_w^T + cs_b)
__global__ __launch_bounds__(128)
void attn_kernel(const float* __restrict__ g,      // (B, DIN)
                 const float* __restrict__ gr_w,   // (RC, DIN)
                 const float* __restrict__ gr_b,   // (RC)
                 const float* __restrict__ cs_w,   // (DIN, RC)
                 const float* __restrict__ cs_b,   // (DIN)
                 float* __restrict__ attn)         // (B, DIN)
{
    __shared__ float grs[RC];
    const int b = blockIdx.x, tid = threadIdx.x;
    const float* gb = g + (size_t)b * DIN;
    if (tid < RC) {
        float s = gr_b[tid];
        for (int d = 0; d < DIN; ++d) s += gb[d] * gr_w[(size_t)tid * DIN + d];
        grs[tid] = 0.5f * s * (1.0f + erff(s * 0.70710678118654752f));  // exact gelu
    }
    __syncthreads();
    for (int d = tid; d < DIN; d += 128) {
        float s = cs_b[d];
        #pragma unroll 8
        for (int r = 0; r < RC; ++r) s += grs[r] * cs_w[(size_t)d * RC + r];
        attn[(size_t)b * DIN + d] = sigmoidf_(s);
    }
}

// Stage 4: out_acc (beta? +=, else =) y * attn[b][d]; on the final
// direction also emit the bf16 copy for the out_proj GEMM.
__global__ void gate_accum_kernel(const float* __restrict__ y,
                                  const float* __restrict__ attn,
                                  float* __restrict__ out_acc,
                                  __bf16* __restrict__ out_acc_bf,
                                  int accumulate, int emit_bf)
{
    const size_t i = (size_t)blockIdx.x * blockDim.x + threadIdx.x;
    if (i >= (size_t)MROWS * DIN) return;
    const int b = (int)(i / ((size_t)LL * DIN));
    const int d = (int)(i % DIN);
    float v = y[i] * attn[(size_t)b * DIN + d];
    if (accumulate) v += out_acc[i];
    out_acc[i] = v;
    if (emit_bf) out_acc_bf[i] = (__bf16)v;
}

// =====================================================================
// host-side orchestration
// =====================================================================
extern "C" void kernel_launch(void* const* d_in, const int* in_sizes, int n_in,
                              void* d_out, int out_size, void* d_ws, size_t ws_size,
                              hipStream_t stream)
{
    const float* hidden     = (const float*)d_in[0];   // (B,L,384)
    const float* in_proj_w  = (const float*)d_in[1];   // (1536,384)
    const float* A_log      = (const float*)d_in[2];   // (4,768,16)
    const float* conv_w     = (const float*)d_in[3];   // (4,768,4)
    const float* conv_b     = (const float*)d_in[4];   // (4,768)
    const float* x_proj_w   = (const float*)d_in[5];   // (4,56,768)
    const float* dt_w       = (const float*)d_in[6];   // (4,768,24)
    const float* dt_b       = (const float*)d_in[7];   // (4,768)
    const float* D_param    = (const float*)d_in[8];   // (4,768)
    const float* ln_g       = (const float*)d_in[9];   // (768)
    const float* ln_b       = (const float*)d_in[10];  // (768)
    const float* gr_w       = (const float*)d_in[11];  // (96,768)
    const float* gr_b       = (const float*)d_in[12];  // (96)
    const float* cs_w       = (const float*)d_in[13];  // (768,96)
    const float* cs_b       = (const float*)d_in[14];  // (768)
    const float* out_proj_w = (const float*)d_in[15];  // (384,768)
    float* out = (float*)d_out;                         // (B,L,384)

    // ---------------- workspace carve-up ----------------
    float* ws = (float*)d_ws;
    size_t off = 0;                                      // in floats
    float* xz      = ws + off; off += (size_t)MROWS * (2*DIN);
    float* xconv   = ws + off; off += (size_t)MROWS * DIN;
    float* xdbl    = ws + off; off += (size_t)MROWS * XDBL_W;
    float* delta   = ws + off; off += (size_t)MROWS * DIN;
    float* ydir    = ws + off; off += (size_t)MROWS * DIN;
    float* out_acc = ws + off; off += (size_t)MROWS * DIN;
    float* mu      = ws + off; off += (size_t)MROWS;
    float* rstd    = ws + off; off += (size_t)MROWS;
    float* gvec    = ws + off; off += (size_t)BATCH * DIN;
    float* attn    = ws + off; off += (size_t)BATCH * DIN;
    // bf16 region (2 floats -> 4 bf16; keep 16B alignment via even float offsets)
    __bf16* bws = (__bf16*)(ws + off);
    size_t boff = 0;                                     // in bf16 elements
    __bf16* hidden_bf = bws + boff; boff += (size_t)MROWS * DMODEL;
    __bf16* ipw_bf    = bws + boff; boff += (size_t)(2*DIN) * DMODEL;
    __bf16* opw_bf    = bws + boff; boff += (size_t)DMODEL * DIN;
    __bf16* xconv_bf  = bws + boff; boff += (size_t)MROWS * DIN;
    __bf16* xdbl_bf   = bws + boff; boff += (size_t)MROWS * RRANK_P;
    __bf16* oacc_bf   = bws + boff; boff += (size_t)MROWS * DIN;
    __bf16* xpw_bf    = bws + boff; boff += (size_t)XDBL_P * DIN;
    __bf16* dtw_bf    = bws + boff; boff += (size_t)DIN * RRANK_P;
    (void)ws_size; (void)in_sizes; (void)n_in; (void)out_size;

    #define CASTPAD(src, dst, rs, cs, stride, rd, cd) \
        cast_pad_bf16<<<(unsigned)(((size_t)(rd)*(cd) + 255)/256), 256, 0, stream>>>( \
            src, dst, rs, cs, stride, rd, cd)

    // one-time bf16 operand casts
    CASTPAD(hidden,     hidden_bf, MROWS,  DMODEL, DMODEL, MROWS,  DMODEL);
    CASTPAD(in_proj_w,  ipw_bf,    2*DIN,  DMODEL, DMODEL, 2*DIN,  DMODEL);
    CASTPAD(out_proj_w, opw_bf,    DMODEL, DIN,    DIN,    DMODEL, DIN);

    // 1) in_proj: xz = hidden @ in_proj_w^T   M=25088, N=1536, K=384
    {
        dim3 grid((2*DIN)/64, MROWS/16);
        gemm_wmma<<<grid, 32, 0, stream>>>(hidden_bf, DMODEL, ipw_bf, DMODEL,
                                           xz, 2*DIN, 2*DIN, DMODEL);
    }

    for (int dir = 0; dir < NDIR; ++dir) {
        const float* cw  = conv_w   + (size_t)dir * DIN * KCONV;
        const float* cb  = conv_b   + (size_t)dir * DIN;
        const float* xpw = x_proj_w + (size_t)dir * XDBL_W * DIN;
        const float* dtw = dt_w     + (size_t)dir * DIN * RRANK;
        const float* dtb = dt_b     + (size_t)dir * DIN;
        const float* al  = A_log    + (size_t)dir * DIN * NST;
        const float* dp  = D_param  + (size_t)dir * DIN;

        // cast + zero-pad the small weights (N -> mult 64, K -> mult 32)
        CASTPAD(xpw, xpw_bf, XDBL_W, DIN,   DIN,   XDBL_P, DIN);
        CASTPAD(dtw, dtw_bf, DIN,    RRANK, RRANK, DIN,    RRANK_P);

        // 2) depthwise causal conv + SiLU (direction-mapped gather)
        {
            dim3 grid((LL*DIN + 255)/256, BATCH);
            conv_silu_kernel<<<grid, 256, 0, stream>>>(xz, cw, cb, xconv, xconv_bf, dir);
        }
        // 3) x_proj: xdbl = xconv @ x_proj_w^T   N=56 (padded 64), K=768
        {
            dim3 grid(XDBL_P/64, MROWS/16);
            gemm_wmma<<<grid, 32, 0, stream>>>(xconv_bf, DIN, xpw_bf, DIN,
                                               xdbl, XDBL_W, XDBL_W, DIN);
        }
        // cast dt slice of xdbl (cols 0..23) with zero pad to K=32
        CASTPAD(xdbl, xdbl_bf, MROWS, RRANK, XDBL_W, MROWS, RRANK_P);
        // 4) dt projection: delta = xdbl[:, :24] @ dt_w^T   N=768, K=24 (pad 32)
        {
            dim3 grid(DIN/64, MROWS/16);
            gemm_wmma<<<grid, 32, 0, stream>>>(xdbl_bf, RRANK_P, dtw_bf, RRANK_P,
                                               delta, DIN, DIN, RRANK_P);
        }
        // 5) selective scan + gate, writes ydir in spatial order
        {
            dim3 grid(DIN/256, BATCH);
            scan_kernel<<<grid, 256, 0, stream>>>(xconv, delta, xdbl, xz,
                                                  al, dtb, dp, ydir, dir);
        }
        // 6) bi-attention
        {
            dim3 grid(LL, BATCH);
            ln_stats_kernel<<<grid, 256, 0, stream>>>(ydir, mu, rstd);
        }
        {
            dim3 grid(DIN/256, BATCH);
            ln_mean_kernel<<<grid, 256, 0, stream>>>(ydir, mu, rstd, ln_g, ln_b, gvec);
        }
        attn_kernel<<<BATCH, 128, 0, stream>>>(gvec, gr_w, gr_b, cs_w, cs_b, attn);
        {
            size_t tot = (size_t)MROWS * DIN;
            dim3 grid((unsigned)((tot + 255)/256));
            gate_accum_kernel<<<grid, 256, 0, stream>>>(ydir, attn, out_acc, oacc_bf,
                                                        dir == 0 ? 0 : 1,
                                                        dir == NDIR-1 ? 1 : 0);
        }
    }

    // 7) out_proj: out = out_acc @ out_proj_w^T   N=384, K=768
    {
        dim3 grid(DMODEL/64, MROWS/16);
        gemm_wmma<<<grid, 32, 0, stream>>>(oacc_bf, DIN, opw_bf, DIN,
                                           out, DMODEL, DMODEL, DIN);
    }
    #undef CASTPAD
}
